// SigmoidAttention_33818572489387
// MI455X (gfx1250) — compile-verified, matrix operand order checked by reading, not run on hardware
//
#include <hip/hip_runtime.h>
#include <hip/hip_bf16.h>

// ---------------------------------------------------------------------------
// Sigmoid attention, MI455X (gfx1250, wave32, WMMA + async-to-LDS).
// B=2, N=2048, DIM=1024, H=16, Dh=64.  All matmuls via v_wmma_f32_16x16x32_bf16.
// ---------------------------------------------------------------------------

#define B_   2
#define N_   2048
#define DIM_ 1024
#define H_   16
#define Dh_  64
#define M_   (B_ * N_)          // 4096 rows of x
#define EPS_ 1e-6f

typedef __attribute__((ext_vector_type(16))) __bf16 v16bf;
typedef __attribute__((ext_vector_type(8)))  float  v8f;
typedef __attribute__((ext_vector_type(4)))  int    v4i;

// ---------------- async global->LDS (CDNA5 ASYNCcnt path), guarded ----------
#if defined(__has_builtin)
#if __has_builtin(__builtin_amdgcn_global_load_async_to_lds_b128) && \
    __has_builtin(__builtin_amdgcn_s_wait_asynccnt)
#define HAVE_ASYNC_LDS 1
#endif
#endif
#ifndef HAVE_ASYNC_LDS
#define HAVE_ASYNC_LDS 0
#endif

#if HAVE_ASYNC_LDS
typedef __attribute__((address_space(1))) v4i gas_v4i;   // global
typedef __attribute__((address_space(3))) v4i las_v4i;   // LDS
#endif

// Copy 16 bytes global -> LDS.  Async variant is tracked by ASYNCcnt.
static __device__ __forceinline__ void copy16_g2l(void* lds_dst, const void* gsrc) {
#if HAVE_ASYNC_LDS
    __builtin_amdgcn_global_load_async_to_lds_b128((gas_v4i*)gsrc, (las_v4i*)lds_dst,
                                                   0, 0);
#else
    *(uint4*)lds_dst = *(const uint4*)gsrc;
#endif
}

static __device__ __forceinline__ void async_wait0() {
#if HAVE_ASYNC_LDS
    __builtin_amdgcn_s_wait_asynccnt(0);
#endif
}

// float -> bf16 with round-to-nearest-even
static __device__ __forceinline__ unsigned short f2bf(float f) {
    unsigned int u = __float_as_uint(f);
    u += 0x7FFFu + ((u >> 16) & 1u);
    return (unsigned short)(u >> 16);
}

// Load a 16x32 bf16 WMMA fragment for this lane.  Per ISA 16-bit A layout:
// lanes 0-15 hold K {0..7, 16..23}, lanes 16-31 hold K {8..15, 24..31}.
// Serves the B operand too when the source is stored [col][k] row-major.
static __device__ __forceinline__ v16bf load_frag(const unsigned short* p, int lane) {
    const unsigned short* q = p + ((lane & 16) ? 8 : 0);
    union { v16bf v; uint4 x[2]; } u;
    u.x[0] = *(const uint4*)(q);
    u.x[1] = *(const uint4*)(q + 16);
    return u.v;
}

static __device__ __forceinline__ v8f wmma_bf16(v16bf a, v16bf b, v8f c) {
    return __builtin_amdgcn_wmma_f32_16x16x32_bf16(false, a, false, b, (short)0, c,
                                                   false, false);
}

// ---------------------------------------------------------------------------
// Elementwise fp32 -> bf16
// ---------------------------------------------------------------------------
__global__ void cvt_bf16_kernel(const float* __restrict__ src,
                                unsigned short* __restrict__ dst, int n) {
    for (int i = blockIdx.x * blockDim.x + threadIdx.x; i < n;
         i += gridDim.x * blockDim.x)
        dst[i] = f2bf(src[i]);
}

// Transposing fp32 [rows][cols] -> bf16 [cols][rows]
__global__ void cvt_bf16_t_kernel(const float* __restrict__ src,
                                  unsigned short* __restrict__ dst,
                                  int rows, int cols) {
    int n = rows * cols;
    for (int i = blockIdx.x * blockDim.x + threadIdx.x; i < n;
         i += gridDim.x * blockDim.x) {
        int r = i / cols, c = i - r * cols;
        dst[c * rows + r] = f2bf(src[i]);
    }
}

// ---------------------------------------------------------------------------
// Tiled GEMM core: 256 threads (8 waves), block tile 128(M) x 128(N),
// K-chunk 32, double-buffered LDS fed by async global->LDS.
// Wave (wm 0..3, wn 0..1) computes 32x64; fragments from ds_load_b128.
// A: [M][K] bf16 row-major.  B: [Ncol][K] bf16 row-major (pre-transposed W).
// ---------------------------------------------------------------------------
#define KSTRIDE 40   // 32 k-elems + 8 pad (80 B rows, 16B aligned)

#define GEMM_STAGE(buf, k0)                                                     \
    {                                                                           \
        _Pragma("unroll") for (int i_ = 0; i_ < 2; ++i_) {                      \
            int c_   = threadIdx.x + 256 * i_;                                  \
            int row_ = c_ >> 2;                                                 \
            int k4_  = (c_ & 3) * 8;                                            \
            copy16_g2l(&As[buf][row_ * KSTRIDE + k4_],                          \
                       ga + (size_t)(am0 + row_) * DIM_ + (k0) + k4_);          \
            copy16_g2l(&Bs[buf][row_ * KSTRIDE + k4_],                          \
                       gb + (size_t)(bn0 + row_) * DIM_ + (k0) + k4_);          \
        }                                                                       \
    }

// ---------------------------------------------------------------------------
// QKV projection: C = Xb @ Wt^T + bias, fused per-head LayerNorm, bf16 output
// scattered to [B,H,N,Dh].  grid = (M/128, 1024/128).
// ---------------------------------------------------------------------------
__global__ void qkv_gemm_kernel(const unsigned short* __restrict__ ga,   // [4096][1024]
                                const unsigned short* __restrict__ gb,   // [1024 n][1024 k]
                                const float* __restrict__ bias,          // [1024]
                                const float* __restrict__ ln_scale,      // [64]
                                const float* __restrict__ ln_bias,       // [64]
                                int do_ln,
                                unsigned short* __restrict__ outh) {     // [B,H,N,Dh]
    __shared__ unsigned short As[2][128 * KSTRIDE];
    __shared__ unsigned short Bs[2][128 * KSTRIDE];

    const int lane = threadIdx.x & 31;
    const int wv   = threadIdx.x >> 5;
    const int wm   = wv >> 1, wn = wv & 1;
    const int lr   = lane & 15;
    const int am0  = blockIdx.x * 128;
    const int bn0  = blockIdx.y * 128;

    v8f acc[2][4] = {};

    GEMM_STAGE(0, 0)
#pragma unroll 1
    for (int kc = 0; kc < DIM_ / 32; ++kc) {
        const int cur = kc & 1;
        async_wait0();
        __syncthreads();                       // chunk kc visible; buffer reuse safe
        if (kc + 1 < DIM_ / 32) GEMM_STAGE((kc + 1) & 1, (kc + 1) * 32)

        v16bf af[2];
#pragma unroll
        for (int mt = 0; mt < 2; ++mt)
            af[mt] = load_frag(&As[cur][(wm * 32 + mt * 16 + lr) * KSTRIDE], lane);
#pragma unroll
        for (int nt = 0; nt < 4; ++nt) {
            v16bf bf = load_frag(&Bs[cur][(wn * 64 + nt * 16 + lr) * KSTRIDE], lane);
            acc[0][nt] = wmma_bf16(af[0], bf, acc[0][nt]);
            acc[1][nt] = wmma_bf16(af[1], bf, acc[1][nt]);
        }
    }

    // epilogue: bias + per-head LayerNorm (64 cols of this wave = one head)
    const int h     = blockIdx.y * 2 + wn;
    const int ncol0 = h * Dh_;
    const int half  = (lane >> 4) & 1;

#pragma unroll
    for (int mt = 0; mt < 2; ++mt) {
#pragma unroll
        for (int nt = 0; nt < 4; ++nt) {
            float bv = bias[ncol0 + nt * 16 + lr];
#pragma unroll
            for (int r = 0; r < 8; ++r) acc[mt][nt][r] += bv;
        }
#pragma unroll
        for (int r = 0; r < 8; ++r) {
            float mean = 0.f, rstd = 0.f;
            if (do_ln) {
                float s  = acc[mt][0][r] + acc[mt][1][r] + acc[mt][2][r] + acc[mt][3][r];
                float s2 = acc[mt][0][r] * acc[mt][0][r] + acc[mt][1][r] * acc[mt][1][r] +
                           acc[mt][2][r] * acc[mt][2][r] + acc[mt][3][r] * acc[mt][3][r];
#pragma unroll
                for (int m = 1; m < 16; m <<= 1) {     // stays within 16-lane halves
                    s  += __shfl_xor(s,  m, 32);
                    s2 += __shfl_xor(s2, m, 32);
                }
                mean = s * (1.0f / Dh_);
                float var = s2 * (1.0f / Dh_) - mean * mean;
                rstd = __frsqrt_rn(var + EPS_);
            }
            const int m  = am0 + wm * 32 + mt * 16 + r + 8 * half;
            const int bb = m >> 11;
            const int ns = m & (N_ - 1);
            unsigned short* orow = outh + (((bb * H_ + h) * N_ + ns) << 6);
#pragma unroll
            for (int nt = 0; nt < 4; ++nt) {
                int d = nt * 16 + lr;
                float y = acc[mt][nt][r];
                if (do_ln) y = (y - mean) * rstd * ln_scale[d] + ln_bias[d];
                orow[d] = f2bf(y);
            }
        }
    }
}

// ---------------------------------------------------------------------------
// Output projection: y = (A @ Wo + bo) * gamma, fp32 out.  grid = (M/128, DIM/128).
// ---------------------------------------------------------------------------
__global__ void out_gemm_kernel(const unsigned short* __restrict__ ga,   // [4096][1024]
                                const unsigned short* __restrict__ gb,   // [1024 n][1024 k]
                                const float* __restrict__ bo,
                                const float* __restrict__ gamma,
                                float* __restrict__ out) {
    __shared__ unsigned short As[2][128 * KSTRIDE];
    __shared__ unsigned short Bs[2][128 * KSTRIDE];

    const int lane = threadIdx.x & 31;
    const int wv   = threadIdx.x >> 5;
    const int wm   = wv >> 1, wn = wv & 1;
    const int lr   = lane & 15;
    const int am0  = blockIdx.x * 128;
    const int bn0  = blockIdx.y * 128;

    v8f acc[2][4] = {};

    GEMM_STAGE(0, 0)
#pragma unroll 1
    for (int kc = 0; kc < DIM_ / 32; ++kc) {
        const int cur = kc & 1;
        async_wait0();
        __syncthreads();
        if (kc + 1 < DIM_ / 32) GEMM_STAGE((kc + 1) & 1, (kc + 1) * 32)

        v16bf af[2];
#pragma unroll
        for (int mt = 0; mt < 2; ++mt)
            af[mt] = load_frag(&As[cur][(wm * 32 + mt * 16 + lr) * KSTRIDE], lane);
#pragma unroll
        for (int nt = 0; nt < 4; ++nt) {
            v16bf bf = load_frag(&Bs[cur][(wn * 64 + nt * 16 + lr) * KSTRIDE], lane);
            acc[0][nt] = wmma_bf16(af[0], bf, acc[0][nt]);
            acc[1][nt] = wmma_bf16(af[1], bf, acc[1][nt]);
        }
    }

    const int half = (lane >> 4) & 1;
#pragma unroll
    for (int mt = 0; mt < 2; ++mt)
#pragma unroll
        for (int nt = 0; nt < 4; ++nt) {
            int n = bn0 + wn * 64 + nt * 16 + lr;
            float bv = bo[n], gv = gamma[n];
#pragma unroll
            for (int r = 0; r < 8; ++r) {
                int m = am0 + wm * 32 + mt * 16 + r + 8 * half;
                out[(size_t)m * DIM_ + n] = (acc[mt][nt][r] + bv) * gv;
            }
        }
}

// ---------------------------------------------------------------------------
// Sigmoid attention per (b, h, 64-row query tile).
// out[i,:] = sum_j sigmoid(q_i . k_j + bias) * v_j    (no normalization)
// grid = (N/64, H, B), block = 128 (4 waves).
// K tile staged via async global->LDS; V staged transposed; P round-trips LDS.
// ---------------------------------------------------------------------------
__global__ void attn_kernel(const unsigned short* __restrict__ qh,  // [B,H,N,Dh]
                            const unsigned short* __restrict__ kh,
                            const unsigned short* __restrict__ vh,
                            const float* __restrict__ init_bias,    // [1]
                            unsigned short* __restrict__ ah) {      // [B,N,H*Dh]
    __shared__ unsigned short Ks[64 * 72];    // K tile:  [j][d], padded stride 72
    __shared__ unsigned short Vt[64 * 72];    // V^T tile: [d][j]
    __shared__ unsigned short Pt[64 * 72];    // P tile:  [row][j]

    const int lane = threadIdx.x & 31;
    const int wv   = threadIdx.x >> 5;
    const int lr   = lane & 15;
    const int half = (lane >> 4) & 1;
    const int it0  = blockIdx.x * 64;
    const int h    = blockIdx.y;
    const int bb   = blockIdx.z;
    const long bh  = (long)(bb * H_ + h) * N_;
    const float b0 = init_bias[0];

    // resident Q fragments for this wave's 16 query rows
    const unsigned short* qrow = qh + (bh + it0 + wv * 16 + lr) * Dh_;
    v16bf qf[2];
    qf[0] = load_frag(qrow, lane);
    qf[1] = load_frag(qrow + 32, lane);

    v8f out[4] = {};

#pragma unroll 1
    for (int j0 = 0; j0 < N_; j0 += 64) {
        __syncthreads();   // previous iteration done reading Ks/Vt/Pt

        // Stage K tile via async: 64 j-rows x 128B, 128 thr x 4 x 16B
#pragma unroll
        for (int itc = 0; itc < 4; ++itc) {
            int c = threadIdx.x + 128 * itc;          // 0..511
            int j = c >> 3;
            int d8 = (c & 7) * 8;
            copy16_g2l(&Ks[j * 72 + d8], kh + (bh + j0 + j) * Dh_ + d8);
        }

        // Stage V^T (transpose needs element scatter -> regs + ds_store)
#pragma unroll
        for (int itc = 0; itc < 4; ++itc) {
            int c = threadIdx.x + 128 * itc;
            int j = c >> 3;
            int db = (c & 7) * 8;
            union { uint4 q; unsigned short s[8]; } u;
            u.q = *(const uint4*)(vh + (bh + j0 + j) * Dh_ + db);
#pragma unroll
            for (int e = 0; e < 8; ++e) Vt[(db + e) * 72 + j] = u.s[e];
        }
        async_wait0();
        __syncthreads();

        // S = Q . K^T  (K fragments from LDS)
        v8f S[4] = {};
#pragma unroll
        for (int nj = 0; nj < 4; ++nj) {
            const unsigned short* krow = &Ks[(nj * 16 + lr) * 72];
            S[nj] = wmma_bf16(qf[0], load_frag(krow, lane),      S[nj]);
            S[nj] = wmma_bf16(qf[1], load_frag(krow + 32, lane), S[nj]);
        }

        // P = sigmoid(S + b0), C-layout -> LDS (A-layout pickup below)
#pragma unroll
        for (int nj = 0; nj < 4; ++nj) {
            int col = nj * 16 + lr;
#pragma unroll
            for (int r = 0; r < 8; ++r) {
                float t = S[nj][r] + b0;
                float p = 1.0f / (1.0f + __expf(-t));
                Pt[(wv * 16 + r + 8 * half) * 72 + col] = f2bf(p);
            }
        }
        __syncthreads();

        // out += P @ V
#pragma unroll
        for (int kb = 0; kb < 2; ++kb) {
            v16bf pf = load_frag(&Pt[(wv * 16 + lr) * 72 + kb * 32], lane);
#pragma unroll
            for (int nd = 0; nd < 4; ++nd) {
                v16bf vf = load_frag(&Vt[(nd * 16 + lr) * 72 + kb * 32], lane);
                out[nd] = wmma_bf16(pf, vf, out[nd]);
            }
        }
    }

    // write to ah[b, n, h*64 + d] (bf16, row-major [4096][1024])
#pragma unroll
    for (int r = 0; r < 8; ++r) {
        int ns = it0 + wv * 16 + r + 8 * half;
        unsigned short* orow = ah + ((long)(bb * N_ + ns) * (H_ * Dh_)) + h * Dh_;
#pragma unroll
        for (int nd = 0; nd < 4; ++nd)
            orow[nd * 16 + lr] = f2bf(out[nd][r]);
    }
}

// ---------------------------------------------------------------------------
// Host side
// ---------------------------------------------------------------------------
extern "C" void kernel_launch(void* const* d_in, const int* in_sizes, int n_in,
                              void* d_out, int out_size, void* d_ws, size_t ws_size,
                              hipStream_t stream) {
    const float* x         = (const float*)d_in[0];
    const float* Wq        = (const float*)d_in[1];
    const float* bq        = (const float*)d_in[2];
    const float* Wk        = (const float*)d_in[3];
    const float* bk        = (const float*)d_in[4];
    const float* Wv        = (const float*)d_in[5];
    const float* bv        = (const float*)d_in[6];
    const float* qn_scale  = (const float*)d_in[7];
    const float* qn_bias   = (const float*)d_in[8];
    const float* kn_scale  = (const float*)d_in[9];
    const float* kn_bias   = (const float*)d_in[10];
    const float* init_bias = (const float*)d_in[11];
    const float* Wo        = (const float*)d_in[12];
    const float* bo        = (const float*)d_in[13];
    const float* gamma     = (const float*)d_in[14];

    char* ws = (char*)d_ws;
    const size_t MiB = 1024 * 1024;
    unsigned short* xb  = (unsigned short*)(ws + 0 * MiB);   // [4096][1024] bf16
    unsigned short* wqt = (unsigned short*)(ws + 8 * MiB);   // [n][k] bf16
    unsigned short* wkt = (unsigned short*)(ws + 10 * MiB);
    unsigned short* wvt = (unsigned short*)(ws + 12 * MiB);
    unsigned short* wot = (unsigned short*)(ws + 14 * MiB);
    unsigned short* qh  = (unsigned short*)(ws + 16 * MiB);  // [B,H,N,Dh] bf16
    unsigned short* kh  = (unsigned short*)(ws + 24 * MiB);
    unsigned short* vh  = (unsigned short*)(ws + 32 * MiB);
    unsigned short* ah  = (unsigned short*)(ws + 40 * MiB);  // [4096][1024] bf16

    cvt_bf16_kernel<<<2048, 256, 0, stream>>>(x, xb, M_ * DIM_);
    cvt_bf16_t_kernel<<<1024, 256, 0, stream>>>(Wq, wqt, DIM_, H_ * Dh_);
    cvt_bf16_t_kernel<<<1024, 256, 0, stream>>>(Wk, wkt, DIM_, H_ * Dh_);
    cvt_bf16_t_kernel<<<1024, 256, 0, stream>>>(Wv, wvt, DIM_, H_ * Dh_);
    cvt_bf16_t_kernel<<<1024, 256, 0, stream>>>(Wo, wot, H_ * Dh_, DIM_);

    dim3 gq(M_ / 128, (H_ * Dh_) / 128);
    qkv_gemm_kernel<<<gq, 256, 0, stream>>>(xb, wqt, bq, qn_scale, qn_bias, 1, qh);
    qkv_gemm_kernel<<<gq, 256, 0, stream>>>(xb, wkt, bk, kn_scale, kn_bias, 1, kh);
    qkv_gemm_kernel<<<gq, 256, 0, stream>>>(xb, wvt, bv, qn_scale, qn_bias, 0, vh);

    dim3 ga(N_ / 64, H_, B_);
    attn_kernel<<<ga, 128, 0, stream>>>(qh, kh, vh, init_bias, ah);

    dim3 go(M_ / 128, DIM_ / 128);
    out_gemm_kernel<<<go, 256, 0, stream>>>(ah, wot, bo, gamma, (float*)d_out);
}
